// FluxonRouterCos_15444702396966
// MI455X (gfx1250) — matrix operand order repaired and checked by reading, not compiled, not run on hardware
//
#include <hip/hip_runtime.h>

typedef unsigned short u16;
typedef __attribute__((ext_vector_type(16))) __bf16 v16bf;
typedef __attribute__((ext_vector_type(8)))  float  v8f;

#define D        512
#define BROWS    4096
#define KROWS    32768
#define KSPLIT   8
#define NSPLIT   (KROWS / KSPLIT)   // 4096 candidates per split
#define WGM      64                 // 64 h-rows per workgroup (4 M-tiles)
#define SLOTS    64                 // KSPLIT * 8 waves partial slots per row

// workspace layout (bytes)
#define A_OFF    0u
#define A_BYTES  (KROWS * D * 2u)               // 33,554,432
#define H_OFF    (A_OFF + A_BYTES)
#define H_BYTES  (BROWS * D * 2u)               // 4,194,304
#define PMAX_OFF (H_OFF + H_BYTES)
#define PMAX_BYTES (BROWS * SLOTS * 4u)         // 1,048,576
#define PIDX_OFF (PMAX_OFF + PMAX_BYTES)

union Frag16 { v16bf v; int4 q[2]; };
union Pack16 { int4 q[2]; u16 u[16]; };

static __device__ __forceinline__ u16 f2bf_rne(float f) {
    unsigned int x = __float_as_uint(f);
    unsigned int r = x + 0x7FFFu + ((x >> 16) & 1u);
    return (u16)(r >> 16);
}

static __device__ __forceinline__ v8f zero8() {
    v8f z = {0.f, 0.f, 0.f, 0.f, 0.f, 0.f, 0.f, 0.f};
    return z;
}

// --- Pass 1/2: L2-normalize rows of [rows, 512] f32 -> bf16 (RNE). One wave per row.
__global__ __launch_bounds__(256) void normalize_rows_bf16(
    const float* __restrict__ src, u16* __restrict__ dst, int rows)
{
    const int row  = blockIdx.x * 8 + (threadIdx.x >> 5);
    const int lane = threadIdx.x & 31;
    if (row >= rows) return;

    const float4* s4 = (const float4*)(src + row * D) + lane * 4;  // 16 floats/lane
    float4 a = s4[0], b = s4[1], c = s4[2], d = s4[3];
    float ss = a.x*a.x + a.y*a.y + a.z*a.z + a.w*a.w
             + b.x*b.x + b.y*b.y + b.z*b.z + b.w*b.w
             + c.x*c.x + c.y*c.y + c.z*c.z + c.w*c.w
             + d.x*d.x + d.y*d.y + d.z*d.z + d.w*d.w;
    #pragma unroll
    for (int m = 1; m < 32; m <<= 1) ss += __shfl_xor(ss, m);
    const float scale = 1.0f / fmaxf(sqrtf(ss), 1e-8f);

    Pack16 p;
    const float v[16] = { a.x,a.y,a.z,a.w, b.x,b.y,b.z,b.w,
                          c.x,c.y,c.z,c.w, d.x,d.y,d.z,d.w };
    #pragma unroll
    for (int i = 0; i < 16; ++i) p.u[i] = f2bf_rne(v[i] * scale);

    int4* o4 = (int4*)(dst + row * D + lane * 16);
    o4[0] = p.q[0];
    o4[1] = p.q[1];
}

// --- Pass 3: fused bf16 WMMA GEMM + running argmax.
// grid = (BROWS/64, KSPLIT), block = 256 (8 waves).
// Per WG: 64 h-rows in LDS (4 M-tiles). Per wave: 32 candidate columns (2 B frags).
// Per 32-K step: 2 global B frags + 4 LDS A frags -> 8 WMMAs (64 FLOP/global byte).
__global__ __launch_bounds__(256) void wmma_argmax(
    const u16* __restrict__ Abf, const u16* __restrict__ Hbf,
    float* __restrict__ pmax, int* __restrict__ pidx)
{
    __shared__ int4 shv[WGM * D / 8];           // 64 KB: 64 h-rows (bf16, row-major)

    const int tid   = threadIdx.x;
    const int wave  = tid >> 5;
    const int lane  = tid & 31;
    const int g     = lane >> 4;                // K-group within fragment
    const int r     = lane & 15;                // row (A-op) / column (B-op) within tile
    const int mBase = blockIdx.x * WGM;
    const int split = blockIdx.y;

    // stage 64 h-rows into LDS (16B chunks)
    const int4* hsrc = (const int4*)(Hbf + mBase * D);
    for (int i = tid; i < WGM * D / 8; i += 256) shv[i] = hsrc[i];
    __syncthreads();

    float rmax[32];
    int   ridx[32];
    #pragma unroll
    for (int i = 0; i < 32; ++i) { rmax[i] = -3.402823466e38f; ridx[i] = 0; }

    const int nEnd = (split + 1) * NSPLIT;
    for (int nBase = split * NSPLIT + wave * 32; nBase < nEnd; nBase += 256) {
        const int n0 = nBase + r;                           // lane's two candidate rows
        const int n1 = nBase + 16 + r;
        const int4* brow0 = (const int4*)Abf + n0 * (D / 8);
        const int4* brow1 = (const int4*)Abf + n1 * (D / 8);

        if (nBase + 256 < nEnd) {                           // global_prefetch_b8
            __builtin_prefetch((const int4*)Abf + (n0 + 256) * (D / 8), 0, 1);
            __builtin_prefetch((const int4*)Abf + (n1 + 256) * (D / 8), 0, 1);
        }

        v8f acc[4][2];
        #pragma unroll
        for (int t = 0; t < 4; ++t) { acc[t][0] = zero8(); acc[t][1] = zero8(); }

        #pragma unroll 2
        for (int kt = 0; kt < D / 32; ++kt) {
            // B fragments (32x16): column n, K = kt*32 + g*16 + {0..15}, contiguous 32B
            const int bi = kt * 4 + g * 2;
            Frag16 bf0, bf1;
            bf0.q[0] = brow0[bi];  bf0.q[1] = brow0[bi + 1];
            bf1.q[0] = brow1[bi];  bf1.q[1] = brow1[bi + 1];

            // A fragments (16x32) from LDS: row = t*16 + r,
            // K = kt*32 + g*16 + {0..15}, contiguous 32B
            Frag16 hf[4];
            #pragma unroll
            for (int t = 0; t < 4; ++t) {
                const int hbase = (t * 16 + r) * (D / 8) + kt * 4 + g * 2;
                hf[t].q[0] = shv[hbase];
                hf[t].q[1] = shv[hbase + 1];
            }

            #pragma unroll
            for (int t = 0; t < 4; ++t) {
                acc[t][0] = __builtin_amdgcn_wmma_f32_16x16x32_bf16(
                                false, hf[t].v, false, bf0.v, (short)0, acc[t][0],
                                false, false);
                acc[t][1] = __builtin_amdgcn_wmma_f32_16x16x32_bf16(
                                false, hf[t].v, false, bf1.v, (short)0, acc[t][1],
                                false, false);
            }
        }

        // running argmax: slot s = t*8 + j holds (M = t*16 + j (+8 for high half))
        // update n0 column before n1 (ascending -> first-max semantics with strict >)
        #pragma unroll
        for (int t = 0; t < 4; ++t) {
            #pragma unroll
            for (int j = 0; j < 8; ++j) {
                const int s = t * 8 + j;
                const float v0 = acc[t][0][j];
                if (v0 > rmax[s]) { rmax[s] = v0; ridx[s] = n0; }
                const float v1 = acc[t][1][j];
                if (v1 > rmax[s]) { rmax[s] = v1; ridx[s] = n1; }
            }
        }
    }

    // argmax across the 16 N-lanes of each half (C-layout: lanes 0-15 / 16-31)
    #pragma unroll
    for (int s = 0; s < 32; ++s) {
        float mx = rmax[s]; int ix = ridx[s];
        #pragma unroll
        for (int m = 1; m < 16; m <<= 1) {
            const float om = __shfl_xor(mx, m);
            const int   oi = __shfl_xor(ix, m);
            if (om > mx || (om == mx && oi < ix)) { mx = om; ix = oi; }
        }
        rmax[s] = mx; ridx[s] = ix;
    }

    if (r == 0) {  // lanes 0 (M = j) and 16 (M = j+8) hold the reduced results
        const int slot = split * 8 + wave;  // 64 partial slots per row
        #pragma unroll
        for (int t = 0; t < 4; ++t) {
            #pragma unroll
            for (int j = 0; j < 8; ++j) {
                const int row = mBase + t * 16 + j + g * 8;
                pmax[row * SLOTS + slot] = rmax[t * 8 + j];
                pidx[row * SLOTS + slot] = ridx[t * 8 + j];
            }
        }
    }
}

// --- Pass 4: combine 64 partial (max, idx) per row, write int argmax. Wave per row.
__global__ __launch_bounds__(256) void final_argmax(
    const float* __restrict__ pmax, const int* __restrict__ pidx,
    int* __restrict__ out)
{
    const int row  = blockIdx.x * 8 + (threadIdx.x >> 5);
    const int lane = threadIdx.x & 31;
    float mx = pmax[row * SLOTS + lane];
    int   ix = pidx[row * SLOTS + lane];
    const float m2 = pmax[row * SLOTS + 32 + lane];
    const int   i2 = pidx[row * SLOTS + 32 + lane];
    if (m2 > mx || (m2 == mx && i2 < ix)) { mx = m2; ix = i2; }
    #pragma unroll
    for (int m = 1; m < 32; m <<= 1) {
        const float om = __shfl_xor(mx, m);
        const int   oi = __shfl_xor(ix, m);
        if (om > mx || (om == mx && oi < ix)) { mx = om; ix = oi; }
    }
    if (lane == 0) out[row] = ix;
}

extern "C" void kernel_launch(void* const* d_in, const int* in_sizes, int n_in,
                              void* d_out, int out_size, void* d_ws, size_t ws_size,
                              hipStream_t stream)
{
    const float* h = (const float*)d_in[0];   // [4096, 512]
    const float* A = (const float*)d_in[1];   // [32768, 512]

    char* ws    = (char*)d_ws;
    u16*  Abf   = (u16*)(ws + A_OFF);
    u16*  Hbf   = (u16*)(ws + H_OFF);
    float* pmax = (float*)(ws + PMAX_OFF);
    int*   pidx = (int*)(ws + PIDX_OFF);

    normalize_rows_bf16<<<KROWS / 8, 256, 0, stream>>>(A, Abf, KROWS);
    normalize_rows_bf16<<<BROWS / 8, 256, 0, stream>>>(h, Hbf, BROWS);

    dim3 grid(BROWS / WGM, KSPLIT);           // 64 x 8 = 512 workgroups
    wmma_argmax<<<grid, 256, 0, stream>>>(Abf, Hbf, pmax, pidx);

    final_argmax<<<BROWS / 8, 256, 0, stream>>>(pmax, pidx, (int*)d_out);
}
